// SpectralFilterModule_62277025792775
// MI455X (gfx1250) — matrix operand order repaired and checked by reading, not compiled, not run on hardware
//
#include <hip/hip_runtime.h>
#include <hip/hip_bf16.h>

typedef _Float16 half16 __attribute__((ext_vector_type(16)));
typedef _Float16 half8  __attribute__((ext_vector_type(8)));
typedef float    float8 __attribute__((ext_vector_type(8)));

// ---------------------------------------------------------------------------
// WMMA wrapper: D = A(16x32 f16) * B(32x16 f16) + C(16x16 f32)
// ---------------------------------------------------------------------------
__device__ __forceinline__ float8 wmma(half16 a, half16 b, float8 c) {
  return __builtin_amdgcn_wmma_f32_16x16x32_f16(false, a, false, b, (short)0, c,
                                                false, false);
}

// A fragment (16x32 f16): lane l holds row (l&15);
// lanes 0-15: K = {0..15}; lanes 16-31: K = {16..31}.
// Loaded as two 16B chunks from a row-major (K-contiguous) row pointer.
__device__ __forceinline__ half16 load_a_frag(const _Float16* rowptr, int kbase,
                                              int lane) {
  int ko = kbase + ((lane >> 4) << 4);
  half8 lo = *(const half8*)(rowptr + ko);
  half8 hi = *(const half8*)(rowptr + ko + 8);
  half16 r;
#pragma unroll
  for (int i = 0; i < 8; ++i) { r[i] = lo[i]; r[i + 8] = hi[i]; }
  return r;
}

// C/D fragment (16x16 f32): lane l holds column (l&15); VGPR i holds row
// i + 8*(l>>4).  Store to an f16 row-major LDS slab of width W.
__device__ __forceinline__ void store_c_lds(_Float16* sl, float8 c, int j0,
                                            int W, int lane) {
  int j  = j0 + (lane & 15);
  int mb = (lane >> 4) * 8;
#pragma unroll
  for (int i = 0; i < 8; ++i) sl[(mb + i) * W + j] = (_Float16)c[i];
}

__device__ __forceinline__ float gelu_erf(float x) {
  return 0.5f * x * (1.0f + erff(x * 0.70710678118654752440f));
}

// ---------------------------------------------------------------------------
// K0: pack f32 weight [K x N] (row-major) into B-fragment order:
// P[((kt*NT+nt)*32 + lane)*16 + j] = W[kt*32 + (lane>>4)*16 + j][nt*16 + (lane&15)]
// ---------------------------------------------------------------------------
__global__ __launch_bounds__(256) void pack_w(const float* __restrict__ W,
                                              _Float16* __restrict__ P, int K,
                                              int N) {
  int idx = blockIdx.x * 256 + threadIdx.x;
  if (idx >= K * N) return;
  int j    = idx & 15;
  int lane = (idx >> 4) & 31;
  int fi   = idx >> 9;
  int NT   = N >> 4;
  int nt   = fi % NT;
  int kt   = fi / NT;
  int k = kt * 32 + ((lane >> 4) << 4) + j;
  int n = nt * 16 + (lane & 15);
  P[idx] = (_Float16)W[(size_t)k * N + n];
}

// ---------------------------------------------------------------------------
// K1: AvgPool2d(4,4) + f32->f16, x_in[b,128,512,512] -> xh[b, n=16384, c=128]
// block handles (b, hh, ww-chunk of 32); LDS transpose for coalesced writes.
// ---------------------------------------------------------------------------
__global__ __launch_bounds__(256) void pool_kernel(const float* __restrict__ xin,
                                                   _Float16* __restrict__ xh) {
  __shared__ float tile[32][129];
  int blk = blockIdx.x;
  int wc  = blk & 3;
  int hh  = (blk >> 2) & 127;
  int b   = blk >> 9;
  int ww0 = wc * 32;
  for (int p = threadIdx.x; p < 4096; p += 256) {
    int c   = p >> 5;
    int wwL = p & 31;
    const float* base = xin + ((size_t)(b * 128 + c) * 512 + hh * 4) * 512 +
                        (size_t)(ww0 + wwL) * 4;
    float s = 0.f;
#pragma unroll
    for (int r = 0; r < 4; ++r) {
      float4 v = *(const float4*)(base + (size_t)r * 512);
      s += v.x + v.y + v.z + v.w;
    }
    tile[wwL][c] = s * 0.0625f;
  }
  __syncthreads();
  for (int p = threadIdx.x; p < 4096; p += 256) {
    int wwL = p >> 7;
    int c   = p & 127;
    int n   = hh * 128 + ww0 + wwL;
    xh[((size_t)b * 16384 + n) * 128 + c] = (_Float16)tile[wwL][c];
  }
}

// ---------------------------------------------------------------------------
// K2: Q = x@Wq (n x 128), K = x@Wk (n x 32) via WMMA.
// Stores TRANSPOSED: Qt[b][d][n] f16, Kt[b][f][n] f16 (n contiguous).
// Also per-wave-tile sum-of-squares partials (deterministic, no atomics):
//   qkpart[(b*1024+t)*320 + (lane>>4)*160 + col], col 0..127 = Q, 128..159 = K
// ---------------------------------------------------------------------------
__global__ __launch_bounds__(256) void qk_gemm(const _Float16* __restrict__ xh,
                                               const _Float16* __restrict__ pWq,
                                               const _Float16* __restrict__ pWk,
                                               _Float16* __restrict__ Qt,
                                               _Float16* __restrict__ Kt,
                                               float* __restrict__ qkpart) {
  int lane = threadIdx.x & 31;
  int wv   = threadIdx.x >> 5;
  int g    = blockIdx.x * 8 + wv;  // 8192 wave tiles
  int b    = g >> 10;
  int t    = g & 1023;
  int n0   = t * 16;

  const _Float16* xrow =
      xh + ((size_t)b * 16384 + n0 + (lane & 15)) * 128;
  half16 a[4];
#pragma unroll
  for (int kt = 0; kt < 4; ++kt) a[kt] = load_a_frag(xrow, kt * 32, lane);

  float* part = qkpart + (size_t)g * 320 + ((lane >> 4) * 160) + (lane & 15);

  // Q: 8 n-tiles of 16
#pragma unroll
  for (int nt = 0; nt < 8; ++nt) {
    float8 c = {};
#pragma unroll
    for (int kt = 0; kt < 4; ++kt) {
      half16 bf = *(const half16*)(pWq + (((kt * 8 + nt) * 32 + lane) << 4));
      c = wmma(a[kt], bf, c);
    }
    half8 hv;
    float ss = 0.f;
#pragma unroll
    for (int i = 0; i < 8; ++i) { ss += c[i] * c[i]; hv[i] = (_Float16)c[i]; }
    size_t dst = ((size_t)b * 128 + nt * 16 + (lane & 15)) * 16384 + n0 +
                 ((lane >> 4) << 3);
    *(half8*)(Qt + dst) = hv;
    part[nt * 16]       = ss;
  }
  // K: 2 n-tiles of 16
#pragma unroll
  for (int nt = 0; nt < 2; ++nt) {
    float8 c = {};
#pragma unroll
    for (int kt = 0; kt < 4; ++kt) {
      half16 bf = *(const half16*)(pWk + (((kt * 2 + nt) * 32 + lane) << 4));
      c = wmma(a[kt], bf, c);
    }
    half8 hv;
    float ss = 0.f;
#pragma unroll
    for (int i = 0; i < 8; ++i) { ss += c[i] * c[i]; hv[i] = (_Float16)c[i]; }
    size_t dst = ((size_t)b * 32 + nt * 16 + (lane & 15)) * 16384 + n0 +
                 ((lane >> 4) << 3);
    *(half8*)(Kt + dst) = hv;
    part[128 + nt * 16] = ss;
  }
}

// ---------------------------------------------------------------------------
// K3: attn_raw[f,d] = sum_n K[n,f] * Q[n,d]  (K-split over n, WMMA).
// grid = b(8) x chunk(32); chunk covers 512 n.  Partials stored per chunk.
// ---------------------------------------------------------------------------
__global__ __launch_bounds__(256) void attn_gemm(const _Float16* __restrict__ Qt,
                                                 const _Float16* __restrict__ Kt,
                                                 float* __restrict__ attn_part) {
  int lane = threadIdx.x & 31;
  int wv   = threadIdx.x >> 5;
  int b    = blockIdx.x >> 5;
  int ck   = blockIdx.x & 31;
  int kb   = ck * 512;
  int mt   = wv >> 2;        // f-tile 0..1
  int ntb  = (wv & 3) * 2;   // d-tiles {ntb, ntb+1}

  const _Float16* Arow =
      Kt + ((size_t)b * 32 + mt * 16 + (lane & 15)) * 16384;
  const _Float16* B0 =
      Qt + ((size_t)b * 128 + ntb * 16 + (lane & 15)) * 16384 +
      ((lane >> 4) << 4);
  const _Float16* B1 = B0 + (size_t)16 * 16384;

  float8 c0 = {}, c1 = {};
  for (int kk = 0; kk < 16; ++kk) {
    int off  = kb + kk * 32;
    half16 a = load_a_frag(Arow, off, lane);
    c0 = wmma(a, *(const half16*)(B0 + off), c0);
    c1 = wmma(a, *(const half16*)(B1 + off), c1);
  }
  float* dst = attn_part + (size_t)blockIdx.x * 4096;
#pragma unroll
  for (int i = 0; i < 8; ++i) {
    int f = mt * 16 + ((lane >> 4) << 3) + i;
    dst[f * 128 + ntb * 16 + (lane & 15)]       = c0[i];
    dst[f * 128 + (ntb + 1) * 16 + (lane & 15)] = c1[i];
  }
}

// ---------------------------------------------------------------------------
// K4: reduce norms + attn chunks; normalize by l2 norms (eps 1e-12);
// per-row min-max normalize; write attn f32 to d_out[...] and f16 copy.
// ---------------------------------------------------------------------------
__global__ __launch_bounds__(256) void finalize_attn(
    const float* __restrict__ qkpart, const float* __restrict__ attn_part,
    float* __restrict__ attn_out, _Float16* __restrict__ attn_h) {
  __shared__ float nrm[160];
  __shared__ float at[4096];
  int b = blockIdx.x;
  for (int col = threadIdx.x; col < 160; col += 256) {
    float s        = 0.f;
    const float* p = qkpart + (size_t)b * 1024 * 320 + col;
    for (int t = 0; t < 1024; ++t) { s += p[0] + p[160]; p += 320; }
    nrm[col] = fmaxf(sqrtf(s), 1e-12f);
  }
  __syncthreads();
  for (int e = threadIdx.x; e < 4096; e += 256) {
    float s        = 0.f;
    const float* p = attn_part + (size_t)(b * 32) * 4096 + e;
    for (int c = 0; c < 32; ++c) { s += *p; p += 4096; }
    int f = e >> 7, d = e & 127;
    at[e] = s / (nrm[128 + f] * nrm[d]);
  }
  __syncthreads();
  int lane = threadIdx.x & 31;
  int wv   = threadIdx.x >> 5;
#pragma unroll
  for (int r = 0; r < 4; ++r) {
    int f = wv * 4 + r;
    float v[4];
#pragma unroll
    for (int q = 0; q < 4; ++q) v[q] = at[f * 128 + q * 32 + lane];
    float mn = fminf(fminf(v[0], v[1]), fminf(v[2], v[3]));
    float mx = fmaxf(fmaxf(v[0], v[1]), fmaxf(v[2], v[3]));
#pragma unroll
    for (int s = 16; s >= 1; s >>= 1) {
      mn = fminf(mn, __shfl_xor(mn, s, 32));
      mx = fmaxf(mx, __shfl_xor(mx, s, 32));
    }
    float inv = 1.0f / (mx - mn);
#pragma unroll
    for (int q = 0; q < 4; ++q) {
      float o = (v[q] - mn) * inv;
      attn_out[(size_t)b * 4096 + f * 128 + q * 32 + lane] = o;
      attn_h[(size_t)b * 4096 + f * 128 + q * 32 + lane]   = (_Float16)o;
    }
  }
}

// ---------------------------------------------------------------------------
// K5: fused y = x @ attn^T -> MLP1(GELU) -> MLP2(GELU inner) -> out.
// One 16-row n-tile per wave; f32 C -> f16 A via per-wave LDS slab.
// Output written channel-major: out[(b*128+ch)*16384 + n].
// ---------------------------------------------------------------------------
__global__ __launch_bounds__(256) void mlp_fused(
    const _Float16* __restrict__ xh, const _Float16* __restrict__ attn_h,
    const _Float16* __restrict__ pW1a, const float* __restrict__ b1a,
    const _Float16* __restrict__ pW1b, const float* __restrict__ b1b,
    const _Float16* __restrict__ pW2a, const float* __restrict__ b2a,
    const _Float16* __restrict__ pW2b, const float* __restrict__ b2b,
    float* __restrict__ out) {
  __shared__ _Float16 lds[8][16 * 128];
  int lane      = threadIdx.x & 31;
  int wv        = threadIdx.x >> 5;
  _Float16* sl  = lds[wv];
  int g  = blockIdx.x * 8 + wv;
  int b  = g >> 10;
  int n0 = (g & 1023) * 16;

  const _Float16* attnb = attn_h + (size_t)b * 4096;  // [32][128] row-major
  const _Float16* xrow  = xh + ((size_t)b * 16384 + n0 + (lane & 15)) * 128;
  const _Float16* slrow = sl + (lane & 15) * 0;  // (row applied per-width below)

  half16 ax[4];
#pragma unroll
  for (int kt = 0; kt < 4; ++kt) ax[kt] = load_a_frag(xrow, kt * 32, lane);

  float8 acc[8];
  // ---- y = x @ attn^T : N=32 (B column f contiguous over d in attn_h) ----
#pragma unroll
  for (int nt = 0; nt < 2; ++nt) {
    float8 c = {};
#pragma unroll
    for (int kt = 0; kt < 4; ++kt) {
      half16 bf = *(const half16*)(attnb + (nt * 16 + (lane & 15)) * 128 +
                                   kt * 32 + ((lane >> 4) << 4));
      c = wmma(ax[kt], bf, c);
    }
    acc[nt] = c;
  }
#pragma unroll
  for (int nt = 0; nt < 2; ++nt) store_c_lds(sl, acc[nt], nt * 16, 32, lane);
  __syncthreads();
  half16 ay = load_a_frag(sl + (lane & 15) * 32, 0, lane);
  __syncthreads();

  // ---- h1 = gelu(y @ W1a + b1a) : K=32, N=64 ----
#pragma unroll
  for (int nt = 0; nt < 4; ++nt) {
    float8 c = wmma(ay, *(const half16*)(pW1a + ((nt * 32 + lane) << 4)),
                    float8{});
    float bv = b1a[nt * 16 + (lane & 15)];
#pragma unroll
    for (int i = 0; i < 8; ++i) c[i] = gelu_erf(c[i] + bv);
    acc[nt] = c;
  }
#pragma unroll
  for (int nt = 0; nt < 4; ++nt) store_c_lds(sl, acc[nt], nt * 16, 64, lane);
  __syncthreads();
  half16 a1[2];
#pragma unroll
  for (int kt = 0; kt < 2; ++kt)
    a1[kt] = load_a_frag(sl + (lane & 15) * 64, kt * 32, lane);
  __syncthreads();

  // ---- h2 = h1 @ W1b + b1b : K=64, N=128 ----
#pragma unroll
  for (int nt = 0; nt < 8; ++nt) {
    float8 c = {};
#pragma unroll
    for (int kt = 0; kt < 2; ++kt)
      c = wmma(a1[kt],
               *(const half16*)(pW1b + (((kt * 8 + nt) * 32 + lane) << 4)), c);
    float bv = b1b[nt * 16 + (lane & 15)];
#pragma unroll
    for (int i = 0; i < 8; ++i) c[i] += bv;
    acc[nt] = c;
  }
#pragma unroll
  for (int nt = 0; nt < 8; ++nt) store_c_lds(sl, acc[nt], nt * 16, 128, lane);
  __syncthreads();
  half16 a2[4];
#pragma unroll
  for (int kt = 0; kt < 4; ++kt)
    a2[kt] = load_a_frag(sl + (lane & 15) * 128, kt * 32, lane);
  __syncthreads();

  // ---- h3 = gelu(h2 @ W2a + b2a) : K=128, N=128 ----
#pragma unroll
  for (int nt = 0; nt < 8; ++nt) {
    float8 c = {};
#pragma unroll
    for (int kt = 0; kt < 4; ++kt)
      c = wmma(a2[kt],
               *(const half16*)(pW2a + (((kt * 8 + nt) * 32 + lane) << 4)), c);
    float bv = b2a[nt * 16 + (lane & 15)];
#pragma unroll
    for (int i = 0; i < 8; ++i) c[i] = gelu_erf(c[i] + bv);
    acc[nt] = c;
  }
#pragma unroll
  for (int nt = 0; nt < 8; ++nt) store_c_lds(sl, acc[nt], nt * 16, 128, lane);
  __syncthreads();
  half16 a3[4];
#pragma unroll
  for (int kt = 0; kt < 4; ++kt)
    a3[kt] = load_a_frag(sl + (lane & 15) * 128, kt * 32, lane);

  // ---- out = h3 @ W2b + b2b : channel-major store ----
#pragma unroll
  for (int nt = 0; nt < 8; ++nt) {
    float8 c = {};
#pragma unroll
    for (int kt = 0; kt < 4; ++kt)
      c = wmma(a3[kt],
               *(const half16*)(pW2b + (((kt * 8 + nt) * 32 + lane) << 4)), c);
    float bv = b2b[nt * 16 + (lane & 15)];
#pragma unroll
    for (int i = 0; i < 8; ++i) c[i] += bv;
    int ch     = nt * 16 + (lane & 15);
    float* dst = out + ((size_t)b * 128 + ch) * 16384 + n0 + ((lane >> 4) << 3);
    *(float4*)dst       = make_float4(c[0], c[1], c[2], c[3]);
    *(float4*)(dst + 4) = make_float4(c[4], c[5], c[6], c[7]);
  }
  (void)slrow;
}

// ---------------------------------------------------------------------------
extern "C" void kernel_launch(void* const* d_in, const int* in_sizes, int n_in,
                              void* d_out, int out_size, void* d_ws,
                              size_t ws_size, hipStream_t stream) {
  (void)in_sizes; (void)n_in; (void)out_size; (void)ws_size;
  const float* x_in = (const float*)d_in[0];
  const float* Wq   = (const float*)d_in[1];
  const float* Wk   = (const float*)d_in[2];
  const float* W1a  = (const float*)d_in[3];
  const float* b1a  = (const float*)d_in[4];
  const float* W1b  = (const float*)d_in[5];
  const float* b1b  = (const float*)d_in[6];
  const float* W2a  = (const float*)d_in[7];
  const float* b2a  = (const float*)d_in[8];
  const float* W2b  = (const float*)d_in[9];
  const float* b2b  = (const float*)d_in[10];

  float* out_main = (float*)d_out;                          // 8*128*128*128
  float* out_attn = (float*)d_out + (size_t)8 * 128 * 16384;  // 8*32*128

  char* ws   = (char*)d_ws;
  size_t off = 0;
  auto alloc = [&](size_t bytes) {
    void* p = ws + off;
    off += (bytes + 255) & ~(size_t)255;
    return p;
  };
  _Float16* xh        = (_Float16*)alloc((size_t)8 * 16384 * 128 * 2);
  _Float16* Qt        = (_Float16*)alloc((size_t)8 * 128 * 16384 * 2);
  _Float16* Kt        = (_Float16*)alloc((size_t)8 * 32 * 16384 * 2);
  float*    qkpart    = (float*)alloc((size_t)8192 * 320 * 4);
  float*    attn_part = (float*)alloc((size_t)256 * 4096 * 4);
  _Float16* attn_h    = (_Float16*)alloc((size_t)8 * 4096 * 2);
  _Float16* pWq  = (_Float16*)alloc((size_t)128 * 128 * 2);
  _Float16* pWk  = (_Float16*)alloc((size_t)128 * 32 * 2);
  _Float16* pW1a = (_Float16*)alloc((size_t)32 * 64 * 2);
  _Float16* pW1b = (_Float16*)alloc((size_t)64 * 128 * 2);
  _Float16* pW2a = (_Float16*)alloc((size_t)128 * 128 * 2);
  _Float16* pW2b = (_Float16*)alloc((size_t)128 * 128 * 2);

  auto pack = [&](const float* W, _Float16* P, int K, int N) {
    int tot = K * N;
    pack_w<<<(tot + 255) / 256, 256, 0, stream>>>(W, P, K, N);
  };
  pack(Wq, pWq, 128, 128);
  pack(Wk, pWk, 128, 32);
  pack(W1a, pW1a, 32, 64);
  pack(W1b, pW1b, 64, 128);
  pack(W2a, pW2a, 128, 128);
  pack(W2b, pW2b, 128, 128);

  pool_kernel<<<8 * 128 * 4, 256, 0, stream>>>(x_in, xh);
  qk_gemm<<<1024, 256, 0, stream>>>(xh, pWq, pWk, Qt, Kt, qkpart);
  attn_gemm<<<256, 256, 0, stream>>>(Qt, Kt, attn_part);
  finalize_attn<<<8, 256, 0, stream>>>(qkpart, attn_part, out_attn, attn_h);
  mlp_fused<<<1024, 256, 0, stream>>>(xh, attn_h, pW1a, b1a, pW1b, b1b, pW2a,
                                      b2a, pW2b, b2b, out_main);
}
